// EnBaseLayer_2259152797799
// MI455X (gfx1250) — matrix operand order repaired and checked by reading, not compiled, not run on hardware
//
#include <hip/hip_runtime.h>

#define HID  128
#define EF   4
#define NG   20
#define CATK 288        // 280 padded to multiple of 32
#define BE   32         // edges per workgroup
#define BN   32         // nodes per workgroup

typedef __attribute__((ext_vector_type(16))) _Float16 v16h;
typedef __attribute__((ext_vector_type(8)))  _Float16 v8h;
typedef __attribute__((ext_vector_type(8)))  float    v8f;
typedef __attribute__((ext_vector_type(4)))  int      v4i;

// -------- async 16B global->LDS copy (gfx1250 ASYNCcnt path), with fallback --------
#if __has_builtin(__builtin_amdgcn_global_load_async_to_lds_b128)
#define HAVE_ASYNC_LDS 1
#else
#define HAVE_ASYNC_LDS 0
#endif

__device__ __forceinline__ void copy16B_to_lds(const _Float16* g, _Float16* l) {
#if HAVE_ASYNC_LDS
  __builtin_amdgcn_global_load_async_to_lds_b128(
      (__attribute__((address_space(1))) v4i*)g,
      (__attribute__((address_space(3))) v4i*)l, 0, 0);
#else
  *(v8h*)l = *(const v8h*)g;
#endif
}

__device__ __forceinline__ void wait_async_lds() {
#if HAVE_ASYNC_LDS
#if __has_builtin(__builtin_amdgcn_s_wait_asynccnt)
  __builtin_amdgcn_s_wait_asynccnt(0);
#else
  asm volatile("s_wait_asynccnt 0" ::: "memory");
#endif
#endif
}

// ---------------- WMMA fragment loaders (wave32, 16x16x32 f16) ----------------
// A (16x32, f16): lane m = lane&15; halfs 0..7 -> K = kb + 8*laneHi + j,
//                 halfs 8..15 -> K = kb + 16 + 8*laneHi + j   (ISA 7.12.2)
__device__ __forceinline__ v16h ld_frag_a(const _Float16* base, int strideH,
                                          int m, int kb, int laneHi) {
  const v8h* p0 = (const v8h*)(base + m * strideH + kb + 8 * laneHi);
  const v8h* p1 = (const v8h*)(base + m * strideH + kb + 16 + 8 * laneHi);
  v8h a0 = *p0, a1 = *p1;
  return __builtin_shufflevector(a0, a1, 0,1,2,3,4,5,6,7,8,9,10,11,12,13,14,15);
}

// B (32x16, f16): lane n = lane&15; halfs 0..7 -> K = kb + 8*laneHi + j,
// halfs 8..15 -> K = kb + 16 + 8*laneHi + j. BT is N-major: BT[n*kp + k].
__device__ __forceinline__ v16h ld_frag_b(const _Float16* bt, int kp,
                                          int n, int kb, int laneHi) {
  const v8h* p0 = (const v8h*)(bt + n * kp + kb + 8 * laneHi);
  const v8h* p1 = (const v8h*)(bt + n * kp + kb + 16 + 8 * laneHi);
  v8h b0 = *p0, b1 = *p1;
  return __builtin_shufflevector(b0, b1, 0,1,2,3,4,5,6,7,8,9,10,11,12,13,14,15);
}

// One GEMM: out[M x 128] = act(A[M x K] * B[K x 128] + bias). M = mtiles*16.
// 8 waves cover mtiles*8 16x16 tiles. No lane divergence around WMMA (EXEC all-1).
__device__ __forceinline__ void gemm_lds(const _Float16* A, int astride,
                                         const _Float16* BT, int kp,
                                         const float* bias, int ktiles, int mtiles,
                                         _Float16* outH, float* outF, int ostride,
                                         int relu, int wave, int lane) {
  const int laneHi = lane >> 4;
  const int nlane  = lane & 15;
  const int tt = mtiles * 8;
  for (int t = wave; t < tt; t += 8) {
    const int mt = t >> 3;
    const int nt = t & 7;
    v8f acc = {};
    for (int kt = 0; kt < ktiles; ++kt) {
      v16h a = ld_frag_a(A, astride, mt * 16 + nlane, kt * 32, laneHi);
      v16h b = ld_frag_b(BT, kp, nt * 16 + nlane, kt * 32, laneHi);
      acc = __builtin_amdgcn_wmma_f32_16x16x32_f16(
          false, a, false, b, (short)0, acc, false, false);
    }
    const float bv = bias[nt * 16 + nlane];
#pragma unroll
    for (int v = 0; v < 8; ++v) {   // D: VGPR v -> M = v + 8*laneHi, N = nlane
      const int m = mt * 16 + v + 8 * laneHi;
      const int n = nt * 16 + nlane;
      float val = acc[v] + bv;
      if (relu) val = fmaxf(val, 0.0f);
      if (outH) outH[m * ostride + n] = (_Float16)val;
      if (outF) outF[m * ostride + n] = val;
    }
  }
}

// ---------------- prep: fp32 [K x 128] -> fp16 transposed [128 x KP] ----------------
__global__ void prep_w(const float* __restrict__ w, _Float16* __restrict__ wT,
                       int K, int KP, int N) {
  int i = blockIdx.x * blockDim.x + threadIdx.x;
  const int total = N * KP;
  const int stride = gridDim.x * blockDim.x;
  for (; i < total; i += stride) {
    const int n = i / KP, k = i % KP;
    wT[i] = (k < K) ? (_Float16)w[k * N + n] : (_Float16)0.f;
  }
}

// ---------------- prep: h fp32 -> fp16 (once; edge kernel gathers f16) ----------------
__global__ void prep_h16(const float* __restrict__ h, _Float16* __restrict__ h16, long n) {
  long i = (long)blockIdx.x * blockDim.x + threadIdx.x;
  const long stride = (long)gridDim.x * blockDim.x;
  for (; i < n; i += stride) h16[i] = (_Float16)h[i];
}

__global__ void zero_f32(float* __restrict__ p, long n) {
  long i = (long)blockIdx.x * blockDim.x + threadIdx.x;
  const long stride = (long)gridDim.x * blockDim.x;
  for (; i < n; i += stride) p[i] = 0.f;
}

// ---------------- edge kernel: message MLP + gate + coord MLP + scatter ----------------
__global__ __launch_bounds__(256) void egnn_edge_kernel(
    const int* __restrict__ src, const int* __restrict__ dst,
    const float* __restrict__ edge_feat,
    const _Float16* __restrict__ h16, const float* __restrict__ x,
    const _Float16* __restrict__ wt1, const float* __restrict__ eb1,
    const _Float16* __restrict__ wt2, const float* __restrict__ eb2,
    const float* __restrict__ infw, const float* __restrict__ infb,
    const _Float16* __restrict__ wtx, const float* __restrict__ xb1,
    const float* __restrict__ xw2, const float* __restrict__ xb2,
    float* __restrict__ miAcc, float* __restrict__ dxAcc, int nE) {
  __shared__ __align__(16) _Float16 cat[BE][CATK];   // 18.0 KB, reused as hid_f
  __shared__ __align__(16) _Float16 out1[BE][HID];   //  8 KB
  __shared__ __align__(16) _Float16 mijh[BE][HID];   //  8 KB
  __shared__ __align__(16) float    mijf[BE][HID];   // 16 KB
  __shared__ float relx[BE][3];
  __shared__ float eij[BE];
  __shared__ float xs[BE];
  __shared__ int   sdst[BE];
  __shared__ int   ssrc[BE];

  const int tid  = threadIdx.x;
  const int lane = tid & 31;
  const int wave = tid >> 5;
  const int e0   = blockIdx.x * BE;

  // ---- stage 1a: per-edge scalar features (rel_x, Gaussian smearing, edge_feat) ----
  if (tid < BE) {
    const int e = tid;
    const int ge = e0 + e;
    int s = 0, d = 0;
    float rx = 0.f, ry = 0.f, rz = 0.f;
    if (ge < nE) {
      s = src[ge]; d = dst[ge];
      rx = x[d * 3 + 0] - x[s * 3 + 0];
      ry = x[d * 3 + 1] - x[s * 3 + 1];
      rz = x[d * 3 + 2] - x[s * 3 + 2];
    }
    ssrc[e] = s; sdst[e] = d;
    relx[e][0] = rx; relx[e][1] = ry; relx[e][2] = rz;
    const float r = rx * rx + ry * ry + rz * rz;
#pragma unroll
    for (int j = 0; j < EF; ++j)
      cat[e][j] = (ge < nE) ? (_Float16)edge_feat[ge * EF + j] : (_Float16)0.f;
    const float spacing = 100.0f / 19.0f;
    const float coeff = -0.5f / (spacing * spacing);
#pragma unroll
    for (int g = 0; g < NG; ++g) {
      const float dr = r - spacing * (float)g;
      cat[e][EF + g] = (_Float16)__expf(coeff * dr * dr);
    }
#pragma unroll
    for (int k = 2 * HID + EF + NG; k < CATK; ++k) cat[e][k] = (_Float16)0.f; // K pad
  }
  __syncthreads();

  // ---- stage 1b: gather h16[dst], h16[src] rows into LDS via async 16B chunks ----
  // 32 chunks per edge (16 per row), BE*32 = 1024 lane-copies -> 4 sweeps of 256.
  for (int idx = tid; idx < BE * 32; idx += 256) {
    const int e     = idx >> 5;
    const int c     = idx & 31;
    const int isSrc = c >> 4;         // 0: h[dst] half, 1: h[src] half
    const int ch    = c & 15;         // 16B chunk within the 128-f16 row
    const int row   = isSrc ? ssrc[e] : sdst[e];
    const _Float16* g = h16 + (size_t)row * HID + ch * 8;
    _Float16* l = &cat[e][EF + NG + isSrc * HID + ch * 8];
    copy16B_to_lds(g, l);
  }
  wait_async_lds();
  __syncthreads();

  // ---- GEMM1: relu(cat @ ew1 + eb1), K = 288 ----
  gemm_lds(&cat[0][0], CATK, wt1, CATK, eb1, CATK / 32, BE / 16,
           &out1[0][0], nullptr, HID, 1, wave, lane);
  __syncthreads();

  // ---- GEMM2: mij = relu(out1 @ ew2 + eb2), K = 128 ----
  gemm_lds(&out1[0][0], HID, wt2, HID, eb2, HID / 32, BE / 16,
           &mijh[0][0], &mijf[0][0], HID, 1, wave, lane);
  __syncthreads();

  // ---- GEMM3: coord hidden = relu(mij @ xw1 + xb1); cat is dead -> reuse as f32 ----
  float* hidf = (float*)&cat[0][0];   // BE*HID*4 = 16 KB <= cat (18 KB)
  gemm_lds(&mijh[0][0], HID, wtx, HID, xb1, HID / 32, BE / 16,
           nullptr, hidf, HID, 1, wave, lane);
  __syncthreads();

  // ---- per-edge reductions: eij = sigmoid(mij.infw + b), x_scale = hid.xw2 + b ----
  if (tid < BE) {
    const int e = tid;
    float a  = infb[0];
    float a2 = xb2[0];
    for (int c = 0; c < HID; ++c) {
      a  += mijf[e][c] * infw[c];
      a2 += hidf[e * HID + c] * xw2[c];
    }
    eij[e] = 1.0f / (1.0f + __expf(-a));
    xs[e] = a2;
  }
  __syncthreads();

  // ---- scatter: mi[dst] += mij*eij ; dx[dst] += rel_x * x_scale ----
  for (int idx = tid; idx < BE * HID; idx += 256) {
    const int e = idx >> 7, c = idx & (HID - 1);
    const int ge = e0 + e;
    if (ge < nE) atomicAdd(&miAcc[sdst[e] * HID + c], mijf[e][c] * eij[e]);
  }
  for (int t = tid; t < BE * 3; t += 256) {
    const int e = t / 3, cmp = t % 3;
    const int ge = e0 + e;
    if (ge < nE) atomicAdd(&dxAcc[sdst[e] * 3 + cmp], relx[e][cmp] * xs[e]);
  }
}

// ---------------- node kernel: h_new = h + MLP([mi,h]); x_new = x + dx ----------------
__global__ __launch_bounds__(256) void egnn_node_kernel(
    const float* __restrict__ h, const _Float16* __restrict__ h16,
    const float* __restrict__ x,
    const float* __restrict__ miAcc, const float* __restrict__ dxAcc,
    const _Float16* __restrict__ wtn1, const float* __restrict__ nb1,
    const _Float16* __restrict__ wtn2, const float* __restrict__ nb2,
    float* __restrict__ hout, float* __restrict__ xout, int nN) {
  __shared__ __align__(16) _Float16 cat2[BN][2 * HID];  // 16 KB, reused as f32 out
  __shared__ __align__(16) _Float16 t1[BN][HID];        //  8 KB

  const int tid  = threadIdx.x;
  const int lane = tid & 31;
  const int wave = tid >> 5;
  const int n0   = blockIdx.x * BN;

  // h half: straight f16 rows, async 16B chunks (BN*16 = 512 copies)
  for (int idx = tid; idx < BN * 16; idx += 256) {
    const int e = idx >> 4, ch = idx & 15;
    const int gn = n0 + e;
    const int row = (gn < nN) ? gn : 0;   // rows for gn>=nN are never stored
    copy16B_to_lds(h16 + (size_t)row * HID + ch * 8, &cat2[e][HID + ch * 8]);
  }
  // mi half: f32 accumulator -> f16
  for (int idx = tid; idx < BN * HID; idx += 256) {
    const int e = idx >> 7, c = idx & (HID - 1);
    const int gn = n0 + e;
    cat2[e][c] = (gn < nN) ? (_Float16)miAcc[(size_t)gn * HID + c] : (_Float16)0.f;
  }
  wait_async_lds();
  __syncthreads();

  gemm_lds(&cat2[0][0], 2 * HID, wtn1, 2 * HID, nb1, (2 * HID) / 32, BN / 16,
           &t1[0][0], nullptr, HID, 1, wave, lane);
  __syncthreads();

  float* o2 = (float*)&cat2[0][0];   // BN*HID*4 = 16 KB == cat2
  gemm_lds(&t1[0][0], HID, wtn2, HID, nb2, HID / 32, BN / 16,
           nullptr, o2, HID, 0, wave, lane);
  __syncthreads();

  for (int idx = tid; idx < BN * HID; idx += 256) {
    const int e = idx >> 7, c = idx & (HID - 1);
    const int gn = n0 + e;
    if (gn < nN) hout[(size_t)gn * HID + c] = h[(size_t)gn * HID + c] + o2[e * HID + c];
  }
  for (int t = tid; t < BN * 3; t += 256) {
    const int e = t / 3, cmp = t % 3;
    const int gn = n0 + e;
    if (gn < nN) xout[gn * 3 + cmp] = x[gn * 3 + cmp] + dxAcc[gn * 3 + cmp];
  }
}

// ---------------- launch ----------------
extern "C" void kernel_launch(void* const* d_in, const int* in_sizes, int n_in,
                              void* d_out, int out_size, void* d_ws, size_t ws_size,
                              hipStream_t stream) {
  const int*   src       = (const int*)d_in[0];
  const int*   dst       = (const int*)d_in[1];
  const float* edge_feat = (const float*)d_in[2];
  const float* h         = (const float*)d_in[3];
  const float* x         = (const float*)d_in[4];
  const float* ew1 = (const float*)d_in[5];
  const float* eb1 = (const float*)d_in[6];
  const float* ew2 = (const float*)d_in[7];
  const float* eb2 = (const float*)d_in[8];
  const float* infw = (const float*)d_in[9];
  const float* infb = (const float*)d_in[10];
  const float* xw1 = (const float*)d_in[11];
  const float* xb1 = (const float*)d_in[12];
  const float* xw2 = (const float*)d_in[13];
  const float* xb2 = (const float*)d_in[14];
  const float* nw1 = (const float*)d_in[15];
  const float* nb1 = (const float*)d_in[16];
  const float* nw2 = (const float*)d_in[17];
  const float* nb2 = (const float*)d_in[18];

  const int nE = in_sizes[0];
  const int nN = in_sizes[3] / HID;

  char* w = (char*)d_ws;
  _Float16* wt1  = (_Float16*)w; w += (size_t)HID * CATK * 2;
  _Float16* wt2  = (_Float16*)w; w += (size_t)HID * HID * 2;
  _Float16* wtx  = (_Float16*)w; w += (size_t)HID * HID * 2;
  _Float16* wtn1 = (_Float16*)w; w += (size_t)HID * 2 * HID * 2;
  _Float16* wtn2 = (_Float16*)w; w += (size_t)HID * HID * 2;
  _Float16* h16  = (_Float16*)w; w += (size_t)nN * HID * 2;
  float* miAcc = (float*)w; w += (size_t)nN * HID * 4;
  float* dxAcc = (float*)w; w += (size_t)nN * 3 * 4;

  prep_w<<<144, 256, 0, stream>>>(ew1, wt1, 2 * HID + EF + NG, CATK, HID);
  prep_w<<<64, 256, 0, stream>>>(ew2, wt2, HID, HID, HID);
  prep_w<<<64, 256, 0, stream>>>(xw1, wtx, HID, HID, HID);
  prep_w<<<128, 256, 0, stream>>>(nw1, wtn1, 2 * HID, 2 * HID, HID);
  prep_w<<<64, 256, 0, stream>>>(nw2, wtn2, HID, HID, HID);
  prep_h16<<<4096, 256, 0, stream>>>(h, h16, (long)nN * HID);
  // miAcc and dxAcc are contiguous in workspace -> single zero pass
  zero_f32<<<2048, 256, 0, stream>>>(miAcc, (long)nN * HID + (long)nN * 3);

  egnn_edge_kernel<<<(nE + BE - 1) / BE, 256, 0, stream>>>(
      src, dst, edge_feat, h16, x, wt1, eb1, wt2, eb2, infw, infb,
      wtx, xb1, xw2, xb2, miAcc, dxAcc, nE);

  float* hout = (float*)d_out;
  float* xout = hout + (size_t)nN * HID;
  egnn_node_kernel<<<(nN + BN - 1) / BN, 256, 0, stream>>>(
      h, h16, x, miAcc, dxAcc, wtn1, nb1, wtn2, nb2, hout, xout, nN);
}
